// RCCAModule_3341484557022
// MI455X (gfx1250) — matrix operand order repaired and checked by reading, not compiled, not run on hardware
//
#include <hip/hip_runtime.h>

typedef __attribute__((ext_vector_type(16))) _Float16 v16h;
typedef __attribute__((ext_vector_type(8)))  _Float16 v8h;
typedef __attribute__((ext_vector_type(8)))  float    v8f;
typedef __attribute__((ext_vector_type(4)))  int      v4i;

#define HW 4096   // 64*64 spatial per (b,c)

#ifndef __has_builtin
#define __has_builtin(x) 0
#endif
#if __has_builtin(__builtin_amdgcn_global_load_async_to_lds_b128)
#define USE_ASYNC_BUILTIN 1
#else
#define USE_ASYNC_BUILTIN 0
#endif
#if __has_builtin(__builtin_amdgcn_s_wait_asynccnt)
#define USE_WAITASYNC_BUILTIN 1
#else
#define USE_WAITASYNC_BUILTIN 0
#endif

// 16-byte global -> LDS async copy (per-lane addresses), tracked by ASYNCcnt.
__device__ inline void async_copy16(const void* g, void* l) {
#if USE_ASYNC_BUILTIN
  __builtin_amdgcn_global_load_async_to_lds_b128(
      (__attribute__((address_space(1))) v4i*)(void*)g,
      (__attribute__((address_space(3))) v4i*)l, 0, 0);
#else
  unsigned lo = (unsigned)(uintptr_t)l;          // LDS aperture: addr[31:0]
  unsigned long long ga = (unsigned long long)(uintptr_t)g;
  asm volatile("global_load_async_to_lds_b128 %0, %1, off"
               :: "v"(lo), "v"(ga) : "memory");
#endif
}
__device__ inline void wait_async0() {
#if USE_WAITASYNC_BUILTIN
  __builtin_amdgcn_s_wait_asynccnt(0);
#else
  asm volatile("s_wait_asynccnt 0" ::: "memory");
#endif
}

// Combine two 8-half LDS chunks into one 16-half WMMA fragment.
__device__ inline v16h frag_from(const _Float16* lo, const _Float16* hi) {
  v8h a = *(const v8h*)lo;
  v8h b = *(const v8h*)hi;
  v16h f;
#pragma unroll
  for (int j = 0; j < 8; ++j) { f[j] = a[j]; f[j + 8] = b[j]; }
  return f;
}

// ---------------------------------------------------------------------------
// Implicit-GEMM conv (3x3 SAME via P=9, or 1x1 via P=1) with WMMA f16->f32,
// double-buffered LDS; weight tiles staged by GLOBAL_LOAD_ASYNC_TO_LDS_B128
// (ASYNCcnt), im2col tiles staged global->VGPR->LDS overlapping the WMMAs.
//   Y[b,co,h,w] = relu?( (sum_k Wp[co,k]*patch[k,(h,w)]) * scale[co] + shift[co] )
//   K order: k = p*Cin + ci  (p = kh*3+kw), Wp prepacked f16 [Co][P*Cin].
// Block: 256 threads = 8 waves as 4(m)x2(n); tile 128(M=Co) x 128(N=spatial).
// ---------------------------------------------------------------------------
__global__ __launch_bounds__(256) void conv_wmma_k(
    const _Float16* __restrict__ Xh, int xC, int xcoff, int Cin,
    const _Float16* __restrict__ Wp, int P,
    const float* __restrict__ scale, const float* __restrict__ shift,
    int Co, int relu, float* __restrict__ Y)
{
  __shared__ _Float16 lA[2][128 * 40];   // [m][k] k-contiguous, pitch 40 halves
  __shared__ _Float16 lB[2][128 * 40];   // [n][k] (B^T) k-contiguous, pitch 40

  const int KT  = P * Cin;
  const int nKB = KT >> 5;
  const int b   = blockIdx.z;
  const int n0  = blockIdx.x * 128;
  const int m0  = blockIdx.y * 128;
  const int h0  = n0 >> 6;            // tile covers rows h0, h0+1, all w
  const int tid = threadIdx.x;
  const int lane = tid & 31, wave = tid >> 5;
  const int wm = wave >> 1, wn = wave & 1;
  const int lr = lane & 15;
  const bool hihalf = (lane >= 16);

  const _Float16* Xb = Xh + (size_t)(b * xC + xcoff) * HW;
  const int ar = tid >> 1, ah = (tid & 1) * 16;  // A loader: row, 16-half chunk
  const int cc = tid >> 3, nb = (tid & 7) * 16;  // B loader: ci lane, n chunk
  const bool aval = (m0 + ar < Co);
  const _Float16* Arow = Wp + (size_t)(m0 + ar) * KT + ah;

  // Rows beyond Co are never overwritten: zero them once in both stages.
  if (!aval) {
    uint4 z = make_uint4(0u, 0u, 0u, 0u);
    *(uint4*)&lA[0][ar * 40 + ah]     = z;
    *(uint4*)&lA[0][ar * 40 + ah + 8] = z;
    *(uint4*)&lA[1][ar * 40 + ah]     = z;
    *(uint4*)&lA[1][ar * 40 + ah + 8] = z;
  }

  v8f acc[2][4] = {};
  _Float16 breg[16];

  // ---- prologue: stage 0
  {
    if (aval) {
      async_copy16(Arow,     &lA[0][ar * 40 + ah]);
      async_copy16(Arow + 8, &lA[0][ar * 40 + ah + 8]);
    }
    const int dh = (P == 9) ? -1 : 0;   // p = 0
    const int dw = (P == 9) ? -1 : 0;
    const _Float16* xc = Xb + (size_t)cc * HW;
#pragma unroll 4
    for (int i = 0; i < 16; ++i) {
      int n = nb + i;
      int h = h0 + (n >> 6) + dh;
      int w = (n & 63) + dw;
      breg[i] = ((unsigned)h < 64u && (unsigned)w < 64u) ? xc[h * 64 + w]
                                                         : (_Float16)0.f;
    }
#pragma unroll 4
    for (int i = 0; i < 16; ++i) lB[0][(nb + i) * 40 + cc] = breg[i];
    wait_async0();
  }
  __syncthreads();

  int p_nx = 0, ci0_nx = 0;            // (p,ci0) of block kb+1, no divisions
  for (int kb = 0; kb < nKB; ++kb) {
    const int s = kb & 1;
    const bool more = (kb + 1 < nKB);

    // ---- prefetch K-block kb+1 into stage s^1 (overlaps the WMMAs below)
    if (more) {
      ci0_nx += 32;
      if (ci0_nx == Cin) { ci0_nx = 0; ++p_nx; }
      const int kg = (kb + 1) << 5;
      if (aval) {
        async_copy16(Arow + kg,     &lA[s ^ 1][ar * 40 + ah]);
        async_copy16(Arow + kg + 8, &lA[s ^ 1][ar * 40 + ah + 8]);
      }
      const int dh = (P == 9) ? (p_nx / 3 - 1) : 0;
      const int dw = (P == 9) ? (p_nx - (p_nx / 3) * 3 - 1) : 0;
      const _Float16* xc = Xb + (size_t)(ci0_nx + cc) * HW;
#pragma unroll 4
      for (int i = 0; i < 16; ++i) {
        int n = nb + i;
        int h = h0 + (n >> 6) + dh;
        int w = (n & 63) + dw;
        breg[i] = ((unsigned)h < 64u && (unsigned)w < 64u) ? xc[h * 64 + w]
                                                           : (_Float16)0.f;
      }
    }

    // ---- fragments per ISA layouts + 2x4 WMMA tiles on stage s
    v16h af[2], bf[4];
#pragma unroll
    for (int t = 0; t < 2; ++t) {
      // A 16x32 f16: lanes0-15 row=lr K{0..7,16..23}; lanes16-31 K{8..15,24..31}
      const _Float16* pa = &lA[s][(wm * 32 + t * 16 + lr) * 40 + (hihalf ? 8 : 0)];
      af[t] = frag_from(pa, pa + 16);
    }
#pragma unroll
    for (int t = 0; t < 4; ++t) {
      // B 32x16 f16: lanes0-15 col=lr K{0..7,16..23}; lanes16-31 K{8..15,24..31}
      const _Float16* pb = &lB[s][(wn * 64 + t * 16 + lr) * 40 + (hihalf ? 8 : 0)];
      bf[t] = frag_from(pb, pb + 16);
    }
#pragma unroll
    for (int tm = 0; tm < 2; ++tm)
#pragma unroll
      for (int tn = 0; tn < 4; ++tn)
        acc[tm][tn] = __builtin_amdgcn_wmma_f32_16x16x32_f16(
            false, af[tm], false, bf[tn], (short)0, acc[tm][tn], false, false);

    // ---- commit prefetched B, drain async A, hand off stages
    if (more) {
#pragma unroll 4
      for (int i = 0; i < 16; ++i) lB[s ^ 1][(nb + i) * 40 + cc] = breg[i];
      wait_async0();
    }
    __syncthreads();
  }

  // ---- epilogue: C/D layout — VGPR r: M = r + (lane<16?0:8), N = lane&15
#pragma unroll
  for (int tm = 0; tm < 2; ++tm) {
    int mb = m0 + wm * 32 + tm * 16 + (hihalf ? 8 : 0);
#pragma unroll
    for (int tn = 0; tn < 4; ++tn) {
      int nn = n0 + wn * 64 + tn * 16 + lr;
#pragma unroll
      for (int r = 0; r < 8; ++r) {
        int m = mb + r;
        if (m < Co) {
          float vv = acc[tm][tn][r] * scale[m] + shift[m];
          if (relu) vv = fmaxf(vv, 0.f);
          Y[((size_t)b * Co + m) * HW + nn] = vv;
        }
      }
    }
  }
}

// ---------------------------------------------------------------------------
// Weight packing: OIHW f32 -> f16 [Co][p][ci]  (k = p*Cin + ci)
// ---------------------------------------------------------------------------
__global__ void pack_w_k(const float* __restrict__ src, _Float16* __restrict__ dst,
                         int Co, int Cin, int P)
{
  size_t o = (size_t)blockIdx.x * 256 + threadIdx.x;
  size_t tot = (size_t)Co * Cin * P;
  if (o >= tot) return;
  int K  = Cin * P;
  int co = (int)(o / K);
  int r  = (int)(o - (size_t)co * K);
  int p  = r / Cin;
  int ci = r - p * Cin;
  dst[o] = (_Float16)src[((size_t)co * Cin + ci) * P + p];
}

// BN / bias folded into per-channel affine (mode 0: BN, 1: bias only, 2: bias then BN)
__global__ void affine_k(float* sc, float* sh, const float* s, const float* b,
                         const float* m, const float* v, const float* bias,
                         int C, int mode)
{
  int c = blockIdx.x * 256 + threadIdx.x;
  if (c >= C) return;
  if (mode == 1) { sc[c] = 1.f; sh[c] = bias[c]; return; }
  float k = s[c] * rsqrtf(v[c] + 1e-5f);
  float t = b[c] - m[c] * k;
  if (mode == 2) t += bias[c] * k;
  sc[c] = k; sh[c] = t;
}

// f32 NCHW -> f16 NCHW (into a buffer with dstC channels at channel offset coff)
__global__ void cvt_h_k(const float* __restrict__ src, _Float16* __restrict__ dst,
                        int C, int dstC, int coff)
{
  size_t idx = (size_t)blockIdx.x * 256 + threadIdx.x;
  size_t per = (size_t)C * HW;
  int b = (int)(idx / per);
  size_t r = idx - (size_t)b * per;
  int c = (int)(r / HW);
  int i = (int)(r - (size_t)c * HW);
  dst[((size_t)b * dstC + coff + c) * HW + i] = (_Float16)src[idx];
}

// ---------------------------------------------------------------------------
// Criss-cross energies + softmax fused: one block(128) per (b,h,w).
//   tid<64: eH over column; tid>=64: eW over row; softmax across 128.
// ---------------------------------------------------------------------------
__global__ __launch_bounds__(128) void cca_att_k(
    const float* __restrict__ q, const float* __restrict__ k,
    float* __restrict__ att)
{
  const int b = blockIdx.y, n = blockIdx.x;
  const int h = n >> 6, w = n & 63;
  const int tid = threadIdx.x, lane = tid & 31, wv = tid >> 5;
  __shared__ float qv[32];
  __shared__ float rmax[4], rsum[4];
  if (tid < 32) qv[tid] = q[((size_t)b * 32 + tid) * HW + n];
  __syncthreads();
  const float* kb_ = k + (size_t)b * 32 * HW;
  float e = 0.f;
  if (tid < 64) {
    int i = tid;
#pragma unroll 8
    for (int c = 0; c < 32; ++c) e += qv[c] * kb_[c * HW + i * 64 + w];
  } else {
    int j = tid - 64;
#pragma unroll 8
    for (int c = 0; c < 32; ++c) e += qv[c] * kb_[c * HW + h * 64 + j];
  }
  float mx = e;
  for (int o = 16; o; o >>= 1) mx = fmaxf(mx, __shfl_xor(mx, o, 32));
  if (lane == 0) rmax[wv] = mx;
  __syncthreads();
  mx = fmaxf(fmaxf(rmax[0], rmax[1]), fmaxf(rmax[2], rmax[3]));
  float ex = __expf(e - mx);
  float sm = ex;
  for (int o = 16; o; o >>= 1) sm += __shfl_xor(sm, o, 32);
  if (lane == 0) rsum[wv] = sm;
  __syncthreads();
  sm = rsum[0] + rsum[1] + rsum[2] + rsum[3];
  att[((size_t)(b * HW) + n) * 128 + tid] = ex / sm;
}

// Aggregation + residual: out = gamma*(oH+oW) + f ; block(256=c) per (b,h,w)
__global__ __launch_bounds__(256) void cca_agg_k(
    const float* __restrict__ att, const float* __restrict__ v,
    const float* __restrict__ f, const float* __restrict__ gamma,
    float* __restrict__ out)
{
  const int b = blockIdx.y, n = blockIdx.x;
  const int h = n >> 6, w = n & 63;
  const int c = threadIdx.x;
  __shared__ float a[128];
  if (c < 128) a[c] = att[((size_t)(b * HW) + n) * 128 + c];
  __syncthreads();
  const float* vb = v + ((size_t)b * 256 + c) * HW;
  float s = 0.f;
#pragma unroll 4
  for (int i = 0; i < 64; ++i) s += a[i] * vb[i * 64 + w];
#pragma unroll 4
  for (int j = 0; j < 64; ++j) s += a[64 + j] * vb[h * 64 + j];
  size_t idx = ((size_t)b * 256 + c) * HW + n;
  out[idx] = gamma[0] * s + f[idx];
}

// Global average pool: one block per (c,b)
__global__ __launch_bounds__(256) void gap_k(const float* __restrict__ fm,
                                             float* __restrict__ vec)
{
  const int c = blockIdx.x, b = blockIdx.y, tid = threadIdx.x;
  __shared__ float r[256];
  const float* p = fm + ((size_t)b * 256 + c) * HW;
  float s = 0.f;
  for (int i = tid; i < HW; i += 256) s += p[i];
  r[tid] = s; __syncthreads();
  for (int o = 128; o; o >>= 1) { if (tid < o) r[tid] += r[tid + o]; __syncthreads(); }
  if (tid == 0) vec[b * 256 + c] = r[0] * (1.f / HW);
}

__global__ void fc1_k(const float* __restrict__ vec, const float* __restrict__ w,
                      const float* __restrict__ bias, float* __restrict__ h)
{
  int b = blockIdx.x, o = threadIdx.x;   // 128
  float s = bias[o];
  for (int c = 0; c < 256; ++c) s += vec[b * 256 + c] * w[o * 256 + c];
  h[b * 128 + o] = fmaxf(s, 0.f);
}

__global__ void fc2_k(const float* __restrict__ h, const float* __restrict__ w,
                      const float* __restrict__ bias, float* __restrict__ gate)
{
  int b = blockIdx.x, o = threadIdx.x;   // 256
  float s = bias[o];
  for (int j = 0; j < 128; ++j) s += h[b * 128 + j] * w[o * 128 + j];
  gate[b * 256 + o] = 1.f / (1.f + __expf(-s));
}

// out_gap = out0 * gate, written as f16 directly into cat channels [2304,2560)
__global__ void gate_apply_k(const float* __restrict__ out0,
                             const float* __restrict__ gate,
                             _Float16* __restrict__ cat)
{
  size_t idx = (size_t)blockIdx.x * 256 + threadIdx.x;   // 8*256*4096 total
  int i = (int)(idx & 4095);
  int c = (int)((idx >> 12) & 255);
  int b = (int)(idx >> 20);
  float vv = out0[idx] * gate[b * 256 + c];
  cat[((size_t)b * 2560 + 2304 + c) * HW + i] = (_Float16)vv;
}

// ---------------------------------------------------------------------------
extern "C" void kernel_launch(void* const* d_in, const int* in_sizes, int n_in,
                              void* d_out, int out_size, void* d_ws, size_t ws_size,
                              hipStream_t stream)
{
  (void)in_sizes; (void)n_in; (void)out_size; (void)ws_size;
  const float* x       = (const float*)d_in[0];
  const float* conva_w = (const float*)d_in[2];
  const float* bn_a_s = (const float*)d_in[3];
  const float* bn_a_b = (const float*)d_in[4];
  const float* bn_a_m = (const float*)d_in[5];
  const float* bn_a_v = (const float*)d_in[6];
  const float* q_w = (const float*)d_in[7];
  const float* q_b = (const float*)d_in[8];
  const float* k_w = (const float*)d_in[9];
  const float* k_b = (const float*)d_in[10];
  const float* v_w = (const float*)d_in[11];
  const float* v_b = (const float*)d_in[12];
  const float* cca_gamma = (const float*)d_in[13];
  const float* convb_w = (const float*)d_in[14];
  const float* bn_b_s = (const float*)d_in[15];
  const float* bn_b_b = (const float*)d_in[16];
  const float* bn_b_m = (const float*)d_in[17];
  const float* bn_b_v = (const float*)d_in[18];
  const float* sq_w = (const float*)d_in[19];
  const float* sq_b = (const float*)d_in[20];
  const float* bn_g_s = (const float*)d_in[21];
  const float* bn_g_b = (const float*)d_in[22];
  const float* bn_g_m = (const float*)d_in[23];
  const float* bn_g_v = (const float*)d_in[24];
  const float* fc1_w = (const float*)d_in[25];
  const float* fc1_b = (const float*)d_in[26];
  const float* fc2_w = (const float*)d_in[27];
  const float* fc2_b = (const float*)d_in[28];
  const float* bott_w = (const float*)d_in[29];
  const float* bn_t_s = (const float*)d_in[30];
  const float* bn_t_b = (const float*)d_in[31];
  const float* bn_t_m = (const float*)d_in[32];
  const float* bn_t_v = (const float*)d_in[33];

  char* ws = (char*)d_ws;
  size_t off = 0;
  auto take = [&](size_t bytes) -> char* {
    char* p = ws + off;
    off = (off + bytes + 255) & ~(size_t)255;
    return p;
  };

  _Float16* cat = (_Float16*)take((size_t)8 * 2560 * HW * 2);
  _Float16* wa  = (_Float16*)take((size_t)256 * 18432 * 2);
  _Float16* wb  = (_Float16*)take((size_t)256 * 2304 * 2);
  _Float16* wt  = (_Float16*)take((size_t)512 * 23040 * 2);
  _Float16* wq  = (_Float16*)take((size_t)32 * 256 * 2);
  _Float16* wk  = (_Float16*)take((size_t)32 * 256 * 2);
  _Float16* wv  = (_Float16*)take((size_t)256 * 256 * 2);
  _Float16* wsq = (_Float16*)take((size_t)256 * 256 * 2);
  float* affA_sc = (float*)take(256 * 4); float* affA_sh = (float*)take(256 * 4);
  float* affB_sc = (float*)take(256 * 4); float* affB_sh = (float*)take(256 * 4);
  float* affT_sc = (float*)take(512 * 4); float* affT_sh = (float*)take(512 * 4);
  float* affG_sc = (float*)take(256 * 4); float* affG_sh = (float*)take(256 * 4);
  float* affQ_sc = (float*)take(32 * 4);  float* affQ_sh = (float*)take(32 * 4);
  float* affK_sc = (float*)take(32 * 4);  float* affK_sh = (float*)take(32 * 4);
  float* affV_sc = (float*)take(256 * 4); float* affV_sh = (float*)take(256 * 4);
  float* out0 = (float*)take((size_t)8 * 256 * HW * 4);
  float* outA = (float*)take((size_t)8 * 256 * HW * 4);
  float* outB = (float*)take((size_t)8 * 256 * HW * 4);
  float* qb   = (float*)take((size_t)8 * 32 * HW * 4);
  float* kbf  = (float*)take((size_t)8 * 32 * HW * 4);
  float* vbf  = (float*)take((size_t)8 * 256 * HW * 4);
  float* att  = (float*)take((size_t)8 * HW * 128 * 4);
  _Float16* fh = (_Float16*)take((size_t)8 * 256 * HW * 2);
  float* vec  = (float*)take(8 * 256 * 4);
  float* hb   = (float*)take(8 * 128 * 4);
  float* gate = (float*)take(8 * 256 * 4);
  float* fm   = outB;   // SE feature map: consumed (GAP) before CCA touches outB

  // ---- weight packing + fused affine params
  pack_w_k<<<18432, 256, 0, stream>>>(conva_w, wa, 256, 2048, 9);
  pack_w_k<<<2304, 256, 0, stream>>>(convb_w, wb, 256, 256, 9);
  pack_w_k<<<46080, 256, 0, stream>>>(bott_w, wt, 512, 2560, 9);
  pack_w_k<<<32, 256, 0, stream>>>(q_w, wq, 32, 256, 1);
  pack_w_k<<<32, 256, 0, stream>>>(k_w, wk, 32, 256, 1);
  pack_w_k<<<256, 256, 0, stream>>>(v_w, wv, 256, 256, 1);
  pack_w_k<<<256, 256, 0, stream>>>(sq_w, wsq, 256, 256, 1);
  affine_k<<<1, 256, 0, stream>>>(affA_sc, affA_sh, bn_a_s, bn_a_b, bn_a_m, bn_a_v, nullptr, 256, 0);
  affine_k<<<1, 256, 0, stream>>>(affB_sc, affB_sh, bn_b_s, bn_b_b, bn_b_m, bn_b_v, nullptr, 256, 0);
  affine_k<<<2, 256, 0, stream>>>(affT_sc, affT_sh, bn_t_s, bn_t_b, bn_t_m, bn_t_v, nullptr, 512, 0);
  affine_k<<<1, 256, 0, stream>>>(affG_sc, affG_sh, bn_g_s, bn_g_b, bn_g_m, bn_g_v, sq_b, 256, 2);
  affine_k<<<1, 256, 0, stream>>>(affQ_sc, affQ_sh, nullptr, nullptr, nullptr, nullptr, q_b, 32, 1);
  affine_k<<<1, 256, 0, stream>>>(affK_sc, affK_sh, nullptr, nullptr, nullptr, nullptr, k_b, 32, 1);
  affine_k<<<1, 256, 0, stream>>>(affV_sc, affV_sh, nullptr, nullptr, nullptr, nullptr, v_b, 256, 1);

  // ---- x -> f16 into cat channels [0,2048)
  cvt_h_k<<<262144, 256, 0, stream>>>(x, cat, 2048, 2560, 0);

  // ---- conva + bn_a  (K = 9*2048 = 18432)
  conv_wmma_k<<<dim3(32, 2, 8), 256, 0, stream>>>(cat, 2560, 0, 2048, wa, 9,
                                                  affA_sc, affA_sh, 256, 0, out0);

  // ---- squeeze-excite path (on out0)
  cvt_h_k<<<32768, 256, 0, stream>>>(out0, fh, 256, 256, 0);
  conv_wmma_k<<<dim3(32, 2, 8), 256, 0, stream>>>(fh, 256, 0, 256, wsq, 1,
                                                  affG_sc, affG_sh, 256, 1, fm);
  gap_k<<<dim3(256, 8), 256, 0, stream>>>(fm, vec);
  fc1_k<<<8, 128, 0, stream>>>(vec, fc1_w, fc1_b, hb);
  fc2_k<<<8, 256, 0, stream>>>(hb, fc2_w, fc2_b, gate);
  gate_apply_k<<<32768, 256, 0, stream>>>(out0, gate, cat);  // cat [2304,2560)

  // ---- CCA x recurrence(=2): ping-pong out0 -> outA -> outB
  float* cur = out0;
  float* nxt = outA;
  for (int it = 0; it < 2; ++it) {
    cvt_h_k<<<32768, 256, 0, stream>>>(cur, fh, 256, 256, 0);
    conv_wmma_k<<<dim3(32, 1, 8), 256, 0, stream>>>(fh, 256, 0, 256, wq, 1,
                                                    affQ_sc, affQ_sh, 32, 0, qb);
    conv_wmma_k<<<dim3(32, 1, 8), 256, 0, stream>>>(fh, 256, 0, 256, wk, 1,
                                                    affK_sc, affK_sh, 32, 0, kbf);
    conv_wmma_k<<<dim3(32, 2, 8), 256, 0, stream>>>(fh, 256, 0, 256, wv, 1,
                                                    affV_sc, affV_sh, 256, 0, vbf);
    cca_att_k<<<dim3(4096, 8), 128, 0, stream>>>(qb, kbf, att);
    cca_agg_k<<<dim3(4096, 8), 256, 0, stream>>>(att, vbf, cur, cca_gamma, nxt);
    cur = nxt;
    nxt = (it == 0) ? outB : out0;
  }

  // ---- convb + bn_b on CCA result (cur == outB), into cat channels [2048,2304)
  cvt_h_k<<<32768, 256, 0, stream>>>(cur, fh, 256, 256, 0);
  conv_wmma_k<<<dim3(32, 2, 8), 256, 0, stream>>>(fh, 256, 0, 256, wb, 9,
                                                  affB_sc, affB_sh, 256, 0, outA);
  cvt_h_k<<<32768, 256, 0, stream>>>(outA, cat, 256, 2560, 2048);

  // ---- bottleneck: 3x3 conv over 2560 channels + bn_t -> d_out (8,512,64,64)
  conv_wmma_k<<<dim3(32, 4, 8), 256, 0, stream>>>(cat, 2560, 0, 2560, wt, 9,
                                                  affT_sc, affT_sh, 512, 0,
                                                  (float*)d_out);
}
